// DeepGMR_33784212750517
// MI455X (gfx1250) — compile-verified, matrix operand order checked by reading, not run on hardware
//
#include <hip/hip_runtime.h>

// ---------------------------------------------------------------------------
// DeepGMR on MI455X (gfx1250): EdgeConv factorized into dense bf16 WMMA GEMMs.
//   y[n,k,o] = LReLU( g[idx[n,k],o] + t[n,o] ),  g = f @ W_lo^T,
//   t = f @ (W_hi - W_lo)^T  -> all matmuls are dense 1024 x O x K, done with
//   v_wmma_f32_16x16x32_bf16 (f32 accumulate).
// kNN reuses the same GEMM:  G = f @ f^T, sq-norms = diag(G), top-20 in regs.
// Main GEMM is register-blocked 32x64 per wave: 8 WMMAs per 6 fragment loads.
// ---------------------------------------------------------------------------

typedef __attribute__((ext_vector_type(16))) __bf16 v16bf;
typedef __attribute__((ext_vector_type(8)))  float  v8f;
typedef unsigned short u16;

#define KNN 20
#define SLOPE 0.2f

__device__ __forceinline__ u16 f2bf(float x) {
    unsigned int u = __float_as_uint(x);
    unsigned int r = u + 0x7FFFu + ((u >> 16) & 1u);   // RNE
    return (u16)(r >> 16);
}
__device__ __forceinline__ float lrelu(float v) { return v > 0.f ? v : SLOPE * v; }

#define WMMA_BF16(a, bfr, c) \
    __builtin_amdgcn_wmma_f32_16x16x32_bf16(false, (a), false, (bfr), (short)0, (c), false, false)

// ---------------------------------------------------------------------------
// Register-blocked GEMM: out[b, M, Nout] = act( A[b, M, K] @ B[b, Nout, K]^T )
// 32x64 tile per wave: 2 A-frags x 4 B-frags -> 8 WMMAs per K-step (0.75
// loads/WMMA instead of 2).  M % 32 == 0, Nout % 64 == 0, K % 32 == 0.
// ---------------------------------------------------------------------------
__global__ void gemm_bf16_wmma_b(const u16* __restrict__ A, long sA,
                                 const u16* __restrict__ Bw, long sB,
                                 float* Cf, u16* Cb, long sC,
                                 int M, int Nout, int Kd, int ldc, int act)
{
    const int b    = blockIdx.y;
    const int Nt   = Nout >> 6;
    const int Mt   = M >> 5;
    const int wave = threadIdx.x >> 5;
    const int tile = blockIdx.x * (blockDim.x >> 5) + wave;
    if (tile >= Mt * Nt) return;                       // uniform per wave
    const int tm = tile / Nt, tn = tile % Nt;
    const int lane = threadIdx.x & 31;
    const int half = lane >> 4, r = lane & 15;

    // k order chosen here (same for A and B): 16 contiguous bf16 per lane; lanes>=16 carry
    // the K=16..31 half.  Row-major storage -> 32B aligned b128 loads.
    const u16* ap = A  + (size_t)b * sA + (size_t)(tm * 32 + r) * Kd + half * 16;
    const u16* bp = Bw + (size_t)b * sB + (size_t)(tn * 64 + r) * Kd + half * 16;
    const size_t sub = (size_t)16 * Kd;                // 16-row/col subtile step

    v8f a00 = {0,0,0,0,0,0,0,0}, a01 = a00, a02 = a00, a03 = a00;
    v8f a10 = a00, a11 = a00, a12 = a00, a13 = a00;

    for (int kk = 0; kk < Kd; kk += 32) {
        __builtin_prefetch(ap + kk + 128, 0, 1);       // global_prefetch_b8
        __builtin_prefetch(bp + kk + 128, 0, 1);
        v16bf fa0 = *reinterpret_cast<const v16bf*>(ap + kk);
        v16bf fa1 = *reinterpret_cast<const v16bf*>(ap + sub + kk);
        v16bf fb0 = *reinterpret_cast<const v16bf*>(bp + kk);
        v16bf fb1 = *reinterpret_cast<const v16bf*>(bp + sub + kk);
        v16bf fb2 = *reinterpret_cast<const v16bf*>(bp + 2 * sub + kk);
        v16bf fb3 = *reinterpret_cast<const v16bf*>(bp + 3 * sub + kk);
        a00 = WMMA_BF16(fa0, fb0, a00);
        a01 = WMMA_BF16(fa0, fb1, a01);
        a02 = WMMA_BF16(fa0, fb2, a02);
        a03 = WMMA_BF16(fa0, fb3, a03);
        a10 = WMMA_BF16(fa1, fb0, a10);
        a11 = WMMA_BF16(fa1, fb1, a11);
        a12 = WMMA_BF16(fa1, fb2, a12);
        a13 = WMMA_BF16(fa1, fb3, a13);
    }

    // C/D layout: lane<16 -> rows v, lane>=16 -> rows 8+v; col = lane&15.
    v8f accs[2][4] = {{a00, a01, a02, a03}, {a10, a11, a12, a13}};
    const int row0 = tm * 32 + half * 8;
    const int col0 = tn * 64 + r;
#pragma unroll
    for (int mi = 0; mi < 2; ++mi)
#pragma unroll
        for (int ni = 0; ni < 4; ++ni)
#pragma unroll
            for (int v = 0; v < 8; ++v) {
                float val = accs[mi][ni][v];
                if (act) val = lrelu(val);
                size_t o = (size_t)b * sC +
                           (size_t)(row0 + mi * 16 + v) * ldc + (col0 + ni * 16);
                if (Cf) Cf[o] = val;
                else    Cb[o] = f2bf(val);
            }
}

// ---------------------------------------------------------------------------
// Simple GEMM (16x16 per wave) for the small logits matmul (Nout = 16).
// ---------------------------------------------------------------------------
__global__ void gemm_bf16_wmma(const u16* __restrict__ A, long sA,
                               const u16* __restrict__ Bw, long sB,
                               float* Cf, u16* Cb, long sC,
                               int M, int Nout, int Kd, int ldc, int act)
{
    const int b    = blockIdx.y;
    const int Nt   = Nout >> 4;
    const int Mt   = M >> 4;
    const int wave = threadIdx.x >> 5;
    const int tile = blockIdx.x * (blockDim.x >> 5) + wave;
    if (tile >= Mt * Nt) return;
    const int tm = tile / Nt, tn = tile % Nt;
    const int lane = threadIdx.x & 31;
    const int half = lane >> 4, r = lane & 15;

    const u16* ap = A  + (size_t)b * sA + (size_t)(tm * 16 + r) * Kd + half * 16;
    const u16* bp = Bw + (size_t)b * sB + (size_t)(tn * 16 + r) * Kd + half * 16;

    v8f acc = {0.f, 0.f, 0.f, 0.f, 0.f, 0.f, 0.f, 0.f};
    for (int kk = 0; kk < Kd; kk += 32) {
        __builtin_prefetch(ap + kk + 128, 0, 1);
        v16bf af = *reinterpret_cast<const v16bf*>(ap + kk);
        v16bf bf = *reinterpret_cast<const v16bf*>(bp + kk);
        acc = WMMA_BF16(af, bf, acc);
    }

    const int row0 = tm * 16 + half * 8;
    const int col  = tn * 16 + r;
#pragma unroll
    for (int v = 0; v < 8; ++v) {
        float val = acc[v];
        if (act) val = lrelu(val);
        size_t o = (size_t)b * sC + (size_t)(row0 + v) * ldc + col;
        if (Cf) Cf[o] = val;
        else    Cb[o] = f2bf(val);
    }
}

// ---------------------------------------------------------------------------
// kNN top-20 from Gram matrix:  d(i,j) = G[i][i] + G[j][j] - 2 G[i][j]
// ---------------------------------------------------------------------------
__global__ void knn_topk(const float* __restrict__ G, long sG,
                         int* __restrict__ idx, int N)
{
    const int b = blockIdx.y;
    const float* Gb = G + (size_t)b * sG;
    __shared__ float sq[1024];
    for (int i = threadIdx.x; i < N; i += blockDim.x)
        sq[i] = Gb[(size_t)i * N + i];
    __syncthreads();

    const int n = blockIdx.x * blockDim.x + threadIdx.x;
    const float* row = Gb + (size_t)n * N;
    const float sqn = sq[n];
    float bd[KNN]; int bi[KNN];
#pragma unroll
    for (int t = 0; t < KNN; ++t) { bd[t] = 3.4e38f; bi[t] = 0; }

    for (int j = 0; j < N; ++j) {
        float d = sqn + sq[j] - 2.f * row[j];
        if (d < bd[KNN - 1]) {
            bd[KNN - 1] = d; bi[KNN - 1] = j;
#pragma unroll
            for (int t = KNN - 1; t > 0; --t) {
                if (bd[t] < bd[t - 1]) {
                    float td = bd[t]; bd[t] = bd[t - 1]; bd[t - 1] = td;
                    int   ti = bi[t]; bi[t] = bi[t - 1]; bi[t - 1] = ti;
                }
            }
        }
    }
    int* op = idx + ((size_t)b * N + n) * KNN;
#pragma unroll
    for (int t = 0; t < KNN; ++t) op[t] = bi[t];
}

// ---------------------------------------------------------------------------
// EdgeConv epilogue: out[n,o] = max_k LReLU(g[idx[n,k],o] + t[n,o]) -> bf16
// gt holds [g | t] as (N, 2O).  Writes layer buffer and/or xcat slice.
// ---------------------------------------------------------------------------
__global__ void gather_max(const float* __restrict__ gt,
                           const int* __restrict__ idx,
                           u16* out1, int ld1, long s1,
                           u16* out2, int ld2, long s2,
                           int O, int N)
{
    const int n = blockIdx.x, b = blockIdx.y, o = threadIdx.x;
    const float* gtb = gt + (size_t)b * N * 2 * O;
    const float  tv  = gtb[(size_t)n * 2 * O + O + o];
    const int*   ip  = idx + ((size_t)b * N + n) * KNN;
    float best = -3.4e38f;
    for (int kk = 0; kk < KNN; ++kk) {
        int m = ip[kk];
        best = fmaxf(best, lrelu(gtb[(size_t)m * 2 * O + o] + tv));
    }
    u16 hv = f2bf(best);
    out1[(size_t)b * s1 + (size_t)n * ld1 + o] = hv;
    if (out2) out2[(size_t)b * s2 + (size_t)n * ld2 + o] = hv;
}

// ---------------------------------------------------------------------------
__global__ void softmax16(const float* __restrict__ logits, float* gamma, int total)
{
    int i = blockIdx.x * blockDim.x + threadIdx.x;
    if (i >= total) return;
    const float* lp = logits + (size_t)i * 16;
    float m = -3.4e38f;
#pragma unroll
    for (int j = 0; j < 16; ++j) m = fmaxf(m, lp[j]);
    float e[16], s = 0.f;
#pragma unroll
    for (int j = 0; j < 16; ++j) { e[j] = __expf(lp[j] - m); s += e[j]; }
    float inv = 1.f / s;
    float* gp = gamma + (size_t)i * 16;
#pragma unroll
    for (int j = 0; j < 16; ++j) gp[j] = e[j] * inv;
}

// ---------------------------------------------------------------------------
// GMM statistics per (b, j):  pi, mu, sigma (+1e-4 I)
// ---------------------------------------------------------------------------
__global__ void gmm_params_kernel(const float* __restrict__ gamma,
                                  const float* __restrict__ pts,
                                  float* pi, float* mu, float* sigma)
{
    const int j = blockIdx.x, b = blockIdx.y, tid = threadIdx.x;
    float s[10];
#pragma unroll
    for (int q = 0; q < 10; ++q) s[q] = 0.f;
    for (int n = tid; n < 1024; n += blockDim.x) {
        float w = gamma[((size_t)b * 1024 + n) * 16 + j];
        const float* p = pts + ((size_t)b * 1024 + n) * 3;
        float x = p[0], y = p[1], z = p[2];
        s[0] += w;
        s[1] += w * x; s[2] += w * y; s[3] += w * z;
        s[4] += w * x * x; s[5] += w * x * y; s[6] += w * x * z;
        s[7] += w * y * y; s[8] += w * y * z; s[9] += w * z * z;
    }
    __shared__ float sm[256][10];
#pragma unroll
    for (int q = 0; q < 10; ++q) sm[tid][q] = s[q];
    __syncthreads();
    for (int st = 128; st > 0; st >>= 1) {
        if (tid < st)
#pragma unroll
            for (int q = 0; q < 10; ++q) sm[tid][q] += sm[tid + st][q];
        __syncthreads();
    }
    if (tid == 0) {
        float S0 = fmaxf(sm[0][0], 1e-12f);
        float inv = 1.f / S0;
        float mx = sm[0][1] * inv, my = sm[0][2] * inv, mz = sm[0][3] * inv;
        int g = b * 16 + j;
        pi[g] = sm[0][0] * (1.f / 1024.f);
        mu[g * 3 + 0] = mx; mu[g * 3 + 1] = my; mu[g * 3 + 2] = mz;
        float* S = sigma + (size_t)g * 9;
        S[0] = sm[0][4] * inv - mx * mx + 1e-4f;
        S[1] = sm[0][5] * inv - mx * my;
        S[2] = sm[0][6] * inv - mx * mz;
        S[3] = S[1];
        S[4] = sm[0][7] * inv - my * my + 1e-4f;
        S[5] = sm[0][8] * inv - my * mz;
        S[6] = S[2];
        S[7] = S[5];
        S[8] = sm[0][9] * inv - mz * mz + 1e-4f;
    }
}

// ---------------------------------------------------------------------------
// Registration: weighted 3x3 correlation -> SVD (Jacobi on A^T A) -> R.
// One lane per batch.  Output: [B,3,3] R then [B,3] zeros (bottom row [:3]).
// ---------------------------------------------------------------------------
__device__ void invert3(const float* m, float inv[3][3])
{
    float a = m[0], bb = m[1], c = m[2], d = m[3], e = m[4], f = m[5],
          g = m[6], h = m[7], i = m[8];
    float A = e * i - f * h, Bc = -(d * i - f * g), Cc = d * h - e * g;
    float det = a * A + bb * Bc + c * Cc;
    float id = 1.f / det;
    inv[0][0] = A * id;  inv[0][1] = -(bb * i - c * h) * id; inv[0][2] = (bb * f - c * e) * id;
    inv[1][0] = Bc * id; inv[1][1] = (a * i - c * g) * id;   inv[1][2] = -(a * f - c * d) * id;
    inv[2][0] = Cc * id; inv[2][1] = -(a * h - bb * g) * id; inv[2][2] = (a * e - bb * d) * id;
}

__global__ void register_kernel(const float* __restrict__ pi,
                                const float* __restrict__ mu_s,
                                const float* __restrict__ mu_t,
                                const float* __restrict__ sigma_t,
                                float* out)
{
    const int b = threadIdx.x;
    if (b >= 8) return;
    float cs[3] = {0, 0, 0}, ct[3] = {0, 0, 0};
    for (int j = 0; j < 16; ++j) {
        float p = pi[b * 16 + j];
        for (int d = 0; d < 3; ++d) {
            cs[d] += p * mu_s[(b * 16 + j) * 3 + d];
            ct[d] += p * mu_t[(b * 16 + j) * 3 + d];
        }
    }
    float Ms[3][3] = {{0,0,0},{0,0,0},{0,0,0}};
    for (int j = 0; j < 16; ++j) {
        float p = pi[b * 16 + j];
        float ds[3], dt[3], inv[3][3], row[3];
        for (int d = 0; d < 3; ++d) {
            ds[d] = mu_s[(b * 16 + j) * 3 + d] - cs[d];
            dt[d] = mu_t[(b * 16 + j) * 3 + d] - ct[d];
        }
        invert3(sigma_t + ((size_t)b * 16 + j) * 9, inv);
        for (int f = 0; f < 3; ++f)
            row[f] = dt[0] * inv[0][f] + dt[1] * inv[1][f] + dt[2] * inv[2][f];
        for (int d = 0; d < 3; ++d)
            for (int f = 0; f < 3; ++f)
                Ms[d][f] += p * ds[d] * row[f];
    }
    float A[3][3];
    for (int d = 0; d < 3; ++d)
        for (int f = 0; f < 3; ++f) {
            float a = Ms[d][f];
            if (!(a == a)) a = 0.f;
            a = fminf(fmaxf(a, -3.0e38f), 3.0e38f);
            A[d][f] = a + 1e-4f;
        }
    float S3[3][3];
    for (int i = 0; i < 3; ++i)
        for (int j = 0; j < 3; ++j)
            S3[i][j] = A[0][i] * A[0][j] + A[1][i] * A[1][j] + A[2][i] * A[2][j];
    float V[3][3] = {{1,0,0},{0,1,0},{0,0,1}};
    for (int sweep = 0; sweep < 8; ++sweep)
        for (int pidx = 0; pidx < 3; ++pidx) {
            int p = (pidx == 2) ? 1 : 0;
            int q = (pidx == 0) ? 1 : 2;
            float apq = S3[p][q];
            if (fabsf(apq) > 1e-20f) {
                float tau = (S3[q][q] - S3[p][p]) / (2.f * apq);
                float t = (tau >= 0.f ? 1.f : -1.f) / (fabsf(tau) + sqrtf(1.f + tau * tau));
                float c = 1.f / sqrtf(1.f + t * t), s = t * c;
                for (int k = 0; k < 3; ++k) {
                    float kp = S3[k][p], kq = S3[k][q];
                    S3[k][p] = c * kp - s * kq; S3[k][q] = s * kp + c * kq;
                }
                for (int k = 0; k < 3; ++k) {
                    float pk = S3[p][k], qk = S3[q][k];
                    S3[p][k] = c * pk - s * qk; S3[q][k] = s * pk + c * qk;
                }
                for (int k = 0; k < 3; ++k) {
                    float kp = V[k][p], kq = V[k][q];
                    V[k][p] = c * kp - s * kq; V[k][q] = s * kp + c * kq;
                }
            }
        }
    float ev[3] = {S3[0][0], S3[1][1], S3[2][2]};
    for (int a = 0; a < 2; ++a)
        for (int c2 = 0; c2 < 2 - a; ++c2)
            if (ev[c2] < ev[c2 + 1]) {
                float t = ev[c2]; ev[c2] = ev[c2 + 1]; ev[c2 + 1] = t;
                for (int k = 0; k < 3; ++k) {
                    float tv = V[k][c2]; V[k][c2] = V[k][c2 + 1]; V[k][c2 + 1] = tv;
                }
            }
    float U[3][3];
    for (int i = 0; i < 3; ++i) {
        float u0 = A[0][0] * V[0][i] + A[0][1] * V[1][i] + A[0][2] * V[2][i];
        float u1 = A[1][0] * V[0][i] + A[1][1] * V[1][i] + A[1][2] * V[2][i];
        float u2 = A[2][0] * V[0][i] + A[2][1] * V[1][i] + A[2][2] * V[2][i];
        float nrm = sqrtf(u0 * u0 + u1 * u1 + u2 * u2);
        float in = nrm > 1e-12f ? 1.f / nrm : 0.f;
        U[0][i] = u0 * in; U[1][i] = u1 * in; U[2][i] = u2 * in;
    }
    float M2[3][3];
    for (int i = 0; i < 3; ++i)
        for (int j = 0; j < 3; ++j)
            M2[i][j] = V[i][0] * U[j][0] + V[i][1] * U[j][1] + V[i][2] * U[j][2];
    float det = M2[0][0] * (M2[1][1] * M2[2][2] - M2[1][2] * M2[2][1])
              - M2[0][1] * (M2[1][0] * M2[2][2] - M2[1][2] * M2[2][0])
              + M2[0][2] * (M2[1][0] * M2[2][1] - M2[1][1] * M2[2][0]);
    for (int i = 0; i < 3; ++i)
        for (int j = 0; j < 3; ++j) {
            float r = V[i][0] * U[j][0] + V[i][1] * U[j][1] + det * V[i][2] * U[j][2];
            out[b * 9 + i * 3 + j] = r;
        }
    for (int i = 0; i < 3; ++i) out[72 + b * 3 + i] = 0.f;   // T[:,3,0:3]
}

// ---------------------------------------------------------------------------
// Prep kernels: f32 -> bf16 weight/feature packing (K padded to mult of 32).
// ---------------------------------------------------------------------------
__global__ void prep_edge_w(const float* __restrict__ W, u16* out, int O, int C, int Kp)
{
    int i = blockIdx.x * blockDim.x + threadIdx.x;
    int tot = 2 * O * Kp;
    if (i >= tot) return;
    int row = i / Kp, c = i % Kp;
    float v = 0.f;
    if (c < C) {
        if (row < O) v = W[row * 2 * C + c];                                 // W_lo
        else { int o = row - O; v = W[o * 2 * C + C + c] - W[o * 2 * C + c]; } // W_hi - W_lo
    }
    out[i] = f2bf(v);
}

__global__ void prep_plain(const float* __restrict__ W, u16* out, int n)
{
    int i = blockIdx.x * blockDim.x + threadIdx.x;
    if (i < n) out[i] = f2bf(W[i]);
}

__global__ void prep_points(const float* __restrict__ pts, float* pts_t, u16* f0, int N)
{
    int i = blockIdx.x * blockDim.x + threadIdx.x;
    if (i >= 8 * N) return;
    int b = i / N, n = i % N;
    float x = pts[(size_t)b * 3 * N + n];
    float y = pts[(size_t)b * 3 * N + N + n];
    float z = pts[(size_t)b * 3 * N + 2 * N + n];
    float* pt = pts_t + (size_t)i * 3;
    pt[0] = x; pt[1] = y; pt[2] = z;
    u16* f = f0 + (size_t)i * 32;
    f[0] = f2bf(x); f[1] = f2bf(y); f[2] = f2bf(z);
    for (int c = 3; c < 32; ++c) f[c] = 0;
}

// ---------------------------------------------------------------------------
extern "C" void kernel_launch(void* const* d_in, const int* in_sizes, int n_in,
                              void* d_out, int out_size, void* d_ws, size_t ws_size,
                              hipStream_t stream)
{
    (void)in_sizes; (void)n_in; (void)out_size; (void)ws_size;
    const float* src = (const float*)d_in[0];
    const float* tgt = (const float*)d_in[1];
    const float* w1  = (const float*)d_in[2];
    const float* w2  = (const float*)d_in[3];
    const float* w3  = (const float*)d_in[4];
    const float* w4  = (const float*)d_in[5];
    const float* w5  = (const float*)d_in[6];
    const float* wc1 = (const float*)d_in[7];
    const float* wc2 = (const float*)d_in[8];
    float* out = (float*)d_out;

    const int N = 1024;
    size_t off = 0;
    auto take = [&](size_t bytes) -> char* {
        off = (off + 255) & ~(size_t)255;
        char* p = (char*)d_ws + off;
        off += bytes;
        return p;
    };

    float* G      = (float*)take(8UL * N * N * 4);        // Gram matrix
    float* gtbuf  = (float*)take(8UL * N * 512 * 4);      // [g | t], max 2O = 512
    u16*   f0     = (u16*)take(8UL * N * 32 * 2);
    u16*   x1     = (u16*)take(8UL * N * 64 * 2);
    u16*   x2     = (u16*)take(8UL * N * 64 * 2);
    u16*   x3     = (u16*)take(8UL * N * 128 * 2);
    u16*   xcat   = (u16*)take(8UL * N * 512 * 2);
    u16*   feat   = (u16*)take(8UL * N * 1024 * 2);
    u16*   hbuf   = (u16*)take(8UL * N * 512 * 2);
    float* logits = (float*)take(8UL * N * 16 * 4);
    float* gammaB[2] = {(float*)take(8UL * N * 16 * 4), (float*)take(8UL * N * 16 * 4)};
    float* ptsT[2]   = {(float*)take(8UL * N * 3 * 4),  (float*)take(8UL * N * 3 * 4)};
    int*   idxb   = (int*)take(8UL * N * KNN * 4);
    u16* wcatA = (u16*)take(2UL * 64 * 32 * 2);
    u16* wcatB = (u16*)take(2UL * 64 * 64 * 2);
    u16* wcatC = (u16*)take(2UL * 128 * 128 * 2);
    u16* wcatD = (u16*)take(2UL * 256 * 256 * 2);
    u16* w5b   = (u16*)take(1024UL * 512 * 2);
    u16* wc1b  = (u16*)take(512UL * 1024 * 2);
    u16* wc2b  = (u16*)take(16UL * 512 * 2);
    float* pi0  = (float*)take(8UL * 16 * 4);
    float* mu0  = (float*)take(8UL * 16 * 3 * 4);
    float* sg0  = (float*)take(8UL * 16 * 9 * 4);
    float* pi1  = (float*)take(8UL * 16 * 4);
    float* mu1  = (float*)take(8UL * 16 * 3 * 4);
    float* sg1  = (float*)take(8UL * 16 * 9 * 4);

    // ---- weight prep (bf16, EdgeConv weights folded into [W_lo ; W_hi-W_lo])
    prep_edge_w<<<(2 * 64 * 32 + 255) / 256, 256, 0, stream>>>(w1, wcatA, 64, 3, 32);
    prep_edge_w<<<(2 * 64 * 64 + 255) / 256, 256, 0, stream>>>(w2, wcatB, 64, 64, 64);
    prep_edge_w<<<(2 * 128 * 128 + 255) / 256, 256, 0, stream>>>(w3, wcatC, 128, 128, 128);
    prep_edge_w<<<(2 * 256 * 256 + 255) / 256, 256, 0, stream>>>(w4, wcatD, 256, 256, 256);
    prep_plain<<<(1024 * 512 + 255) / 256, 256, 0, stream>>>(w5, w5b, 1024 * 512);
    prep_plain<<<(512 * 1024 + 255) / 256, 256, 0, stream>>>(wc1, wc1b, 512 * 1024);
    prep_plain<<<(16 * 512 + 255) / 256, 256, 0, stream>>>(wc2, wc2b, 16 * 512);

    struct Layer { const u16* fin; int Kp; int O; const u16* w; u16* outb; int outld; int xoff; };

    for (int c = 0; c < 2; ++c) {
        const float* pts = c ? tgt : src;
        prep_points<<<(8 * N + 255) / 256, 256, 0, stream>>>(pts, ptsT[c], f0, N);

        Layer Ls[4] = {
            { f0, 32,  64,  wcatA, x1,      64,  0   },
            { x1, 64,  64,  wcatB, x2,      64,  64  },
            { x2, 64,  128, wcatC, x3,      128, 128 },
            { x3, 128, 256, wcatD, nullptr, 0,   256 },
        };

        for (int l = 0; l < 4; ++l) {
            const Layer& L = Ls[l];
            const long sF = (long)N * L.Kp;
            // Gram matrix for kNN:  G = fin @ fin^T   (1024 x 1024 x Kp)
            {
                int tiles = (N / 32) * (N / 64);
                gemm_bf16_wmma_b<<<dim3((tiles + 3) / 4, 8), 128, 0, stream>>>(
                    L.fin, sF, L.fin, sF, G, nullptr, (long)N * N,
                    N, N, L.Kp, N, 0);
            }
            knn_topk<<<dim3(N / 256, 8), 256, 0, stream>>>(G, (long)N * N, idxb, N);
            // [g | t] = fin @ [W_lo ; W_hi-W_lo]^T   (1024 x 2O x Kp)
            const int twoO = 2 * L.O;
            {
                int tiles = (N / 32) * (twoO / 64);
                gemm_bf16_wmma_b<<<dim3((tiles + 3) / 4, 8), 128, 0, stream>>>(
                    L.fin, sF, L.w, 0, gtbuf, nullptr, (long)N * twoO,
                    N, twoO, L.Kp, twoO, 0);
            }
            // max-over-k epilogue
            u16* o1; int ld1; long s1; u16* o2 = nullptr; int ld2 = 0; long s2 = 0;
            if (L.outb) {
                o1 = L.outb; ld1 = L.outld; s1 = (long)N * L.outld;
                o2 = xcat + L.xoff; ld2 = 512; s2 = (long)N * 512;
            } else {
                o1 = xcat + L.xoff; ld1 = 512; s1 = (long)N * 512;
            }
            gather_max<<<dim3(N, 8), L.O, 0, stream>>>(
                gtbuf, idxb, o1, ld1, s1, o2, ld2, s2, L.O, N);
        }

        // feat = LReLU(xcat @ w5^T)   (1024 x 1024 x 512)
        gemm_bf16_wmma_b<<<dim3(((N / 32) * (1024 / 64) + 3) / 4, 8), 128, 0, stream>>>(
            xcat, (long)N * 512, w5b, 0, nullptr, feat, (long)N * 1024,
            N, 1024, 512, 1024, 1);
        // h = LReLU(feat @ wc1^T)     (1024 x 512 x 1024)
        gemm_bf16_wmma_b<<<dim3(((N / 32) * (512 / 64) + 3) / 4, 8), 128, 0, stream>>>(
            feat, (long)N * 1024, wc1b, 0, nullptr, hbuf, (long)N * 512,
            N, 512, 1024, 512, 1);
        // logits = h @ wc2^T          (1024 x 16 x 512) -- small, simple path
        gemm_bf16_wmma<<<dim3((64 + 3) / 4, 8), 128, 0, stream>>>(
            hbuf, (long)N * 512, wc2b, 0, logits, nullptr, (long)N * 16,
            N, 16, 512, 16, 0);
        softmax16<<<(8 * N + 255) / 256, 256, 0, stream>>>(logits, gammaB[c], 8 * N);
    }

    gmm_params_kernel<<<dim3(16, 8), 256, 0, stream>>>(gammaB[0], ptsT[0], pi0, mu0, sg0);
    gmm_params_kernel<<<dim3(16, 8), 256, 0, stream>>>(gammaB[1], ptsT[1], pi1, mu1, sg1);
    register_kernel<<<1, 32, 0, stream>>>(pi0, mu0, mu1, sg1, out);
}
